// CharLSTM_59081570123990
// MI455X (gfx1250) — matrix operand order repaired and checked by reading, not compiled, not run on hardware
//
#include <hip/hip_runtime.h>

#define SEQ    256
#define BATCH  64
#define EMB    512
#define HID    1024
#define VOCAB  128
#define BH     (BATCH * HID)     // 65536
#define LOGITS (BATCH * VOCAB)   // 8192
#define NBLK   (HID / 16)        // 64 persistent blocks
#define KCHUNK 128               // K elements staged in LDS per round
#define LDS_STRIDE 136           // 128 + 8 bf16 pad -> 272B row stride (68 banks)
#define BUFE   (64 * LDS_STRIDE) // elements per LDS buffer (64 rows)

typedef __bf16 bf16_t;
typedef __attribute__((ext_vector_type(16))) __bf16 v16bf;
typedef __attribute__((ext_vector_type(8)))  __bf16 v8bf;
typedef __attribute__((ext_vector_type(8)))  float  v8f;
typedef unsigned int __attribute__((ext_vector_type(4))) u32x4;
typedef int          __attribute__((ext_vector_type(4))) i32x4;
typedef int          __attribute__((ext_vector_type(8))) i32x8;

#ifndef __has_builtin
#define __has_builtin(x) 0
#endif
#if __has_builtin(__builtin_amdgcn_tensor_load_to_lds) && \
    __has_builtin(__builtin_amdgcn_s_wait_tensorcnt)
#define USE_TDM 1
#else
#define USE_TDM 0
#endif

// ---------------------------------------------------------------------------
// WMMA operand layouts as loaded here (contiguous k, wave32):
// A 16x32 bf16: lane L<16 -> row L,    K = kk+[0..15]
//               lane L>=16 -> row L-16, K = kk+16+[0..15]
// B 32x16 bf16: lane L<16 -> col L,    K = kk+[0..15]
//               lane L>=16 -> col L-16, K = kk+16+[0..15]
// C/D: element v -> (m = v + 8*(lane>=16), n = lane%16)
// ---------------------------------------------------------------------------
__device__ __forceinline__ v16bf load_a16(const bf16_t* rowp, int kk, int khalf) {
  union { v16bf v; v8bf h[2]; } u;
  const bf16_t* p = rowp + kk + 2 * khalf;        // khalf = 0 or 8
  u.h[0] = *(const v8bf*)(p);
  u.h[1] = *(const v8bf*)(p + 8);
  return u.v;
}

__device__ __forceinline__ v16bf load_b16_lds(const bf16_t* p) {
  union { v16bf v; v8bf h[2]; } u;
  u.h[0] = *(const v8bf*)(p);
  u.h[1] = *(const v8bf*)(p + 8);
  return u.v;
}

__device__ __forceinline__ v8f wmma_bf16f32(v16bf a, v16bf b, v8f c) {
  return __builtin_amdgcn_wmma_f32_16x16x32_bf16(false, a, false, b,
                                                 (short)0, c, false, false);
}

__device__ __forceinline__ float sigmoidf_(float x) {
  return 1.0f / (1.0f + __expf(-x));
}

// ---------------------------------------------------------------------------
// Device-scope grid barrier (generation counter), agent-scope atomics.
// ---------------------------------------------------------------------------
__device__ __forceinline__ void gsync(unsigned* cnt, unsigned* gen) {
  __threadfence();
  __syncthreads();
  if (threadIdx.x == 0) {
    unsigned g = __hip_atomic_load(gen, __ATOMIC_RELAXED, __HIP_MEMORY_SCOPE_AGENT);
    unsigned a = __hip_atomic_fetch_add(cnt, 1u, __ATOMIC_ACQ_REL, __HIP_MEMORY_SCOPE_AGENT);
    if (a == (unsigned)(gridDim.x - 1)) {
      __hip_atomic_store(cnt, 0u, __ATOMIC_RELAXED, __HIP_MEMORY_SCOPE_AGENT);
      __hip_atomic_fetch_add(gen, 1u, __ATOMIC_RELEASE, __HIP_MEMORY_SCOPE_AGENT);
    } else {
      while (__hip_atomic_load(gen, __ATOMIC_ACQUIRE, __HIP_MEMORY_SCOPE_AGENT) == g)
        __builtin_amdgcn_s_sleep(1);
    }
  }
  __syncthreads();
  __threadfence();
}

#if USE_TDM
// ---------------------------------------------------------------------------
// TDM descriptor (CDNA5 ISA): 2D tile, 16 rows x
// 128 bf16, row stride K elements, LDS padding 4 DWORDs per 64 DWORDs
// (reproduces the 272B padded LDS_STRIDE in hardware).
// ---------------------------------------------------------------------------
__device__ __forceinline__ void tdm_fill_gate(const bf16_t* gptr,  // tile start
                                              int K,               // row stride
                                              bf16_t* ldsdst)      // gate region
{
  unsigned long long ga = (unsigned long long)(uintptr_t)gptr;
  unsigned lds = (unsigned)(uintptr_t)ldsdst;     // low 32 bits = LDS byte addr
  u32x4 g0;
  g0[0] = 1u;                                     // count=1 (valid), user mode
  g0[1] = lds;                                    // lds_addr [63:32]
  g0[2] = (unsigned)(ga & 0xFFFFFFFFu);           // global_addr [95:64]
  g0[3] = (unsigned)((ga >> 32) & 0x01FFFFFFull)  // global_addr [120:96]
          | (2u << 30);                           // type=2 ("image")
  i32x8 g1;
  g1[0] = (1 << 16)    // data_size = 2 bytes
        | (1 << 20)    // pad_enable
        | (5 << 22)    // pad_interval: 64 DWORDs (256B row)
        | (3 << 25);   // pad_amount: 4 DWORDs (16B)
  g1[1] = (int)(((unsigned)K & 0xFFFFu) << 16);   // tensor_dim0[15:0] @ bits 63:48
  g1[2] = (int)(16u << 16);                       // dim0[31:16]=0, tensor_dim1=16
  g1[3] = (int)(128u << 16);                      // dim1[31:16]=0, tile_dim0=128
  g1[4] = 16;                                     // tile_dim1=16, tile_dim2=0
  g1[5] = K;                                      // tensor_dim0_stride[31:0]
  g1[6] = 0;
  g1[7] = 0;
  i32x4 gz = (i32x4)0;
#if __clang_major__ >= 23
  i32x8 gz8 = (i32x8)0;
  __builtin_amdgcn_tensor_load_to_lds(g0, g1, gz, gz, gz8, 0);
#else
  __builtin_amdgcn_tensor_load_to_lds(g0, g1, gz, gz, 0);
#endif
}
#endif // USE_TDM

// ---------------------------------------------------------------------------
// One GEMM contribution: acc[g] += A[64,K] @ W[g*HID + n_base .. +16, :K]^T
// B staged through double-buffered LDS. TDM path: wave0 issues 4 gate-tile
// DMAs for chunk i+1 while all waves consume chunk i; one s_wait_tensorcnt +
// one __syncthreads per chunk. Fallback: cooperative fill via ds_store.
// ---------------------------------------------------------------------------
__device__ __forceinline__ void gemm_part(
    v8f acc[4], const bf16_t* __restrict__ A, int K,
    const bf16_t* __restrict__ W, int n_base, bf16_t* smem)
{
  const int tid   = threadIdx.x;
  const int lane  = tid & 31;
  const int wave  = tid >> 5;
  const int rrow  = lane & 15;
  const int khalf = (lane < 16) ? 0 : 8;
  const int koff  = (lane < 16) ? 0 : 16;

  const bf16_t* arow = A + (size_t)(wave * 16 + rrow) * K;
  const int nchunk = K / KCHUNK;   // 4 or 8 (always even)

#if USE_TDM
  const bf16_t* wg[4];
#pragma unroll
  for (int g = 0; g < 4; ++g)
    wg[g] = W + (size_t)(g * HID + n_base) * K;

  if (wave == 0) {
#pragma unroll
    for (int g = 0; g < 4; ++g)
      tdm_fill_gate(wg[g], K, smem + g * 16 * LDS_STRIDE);
  }

  int cur = 0;
  for (int ci = 0; ci < nchunk; ++ci) {
    if (wave == 0) __builtin_amdgcn_s_wait_tensorcnt(0);
    __syncthreads();                       // buf[cur] ready for all waves
    if (wave == 0 && (ci + 1) < nchunk) {
      bf16_t* nb = smem + (cur ^ 1) * BUFE;
#pragma unroll
      for (int g = 0; g < 4; ++g)
        tdm_fill_gate(wg[g] + (ci + 1) * KCHUNK, K, nb + g * 16 * LDS_STRIDE);
    }
    const bf16_t* buf = smem + cur * BUFE;
#pragma unroll
    for (int kc = 0; kc < 4; ++kc) {
      v16bf a = load_a16(arow, ci * KCHUNK + kc * 32, khalf);
      v16bf bf[4];
#pragma unroll
      for (int g = 0; g < 4; ++g)
        bf[g] = load_b16_lds(buf + (g * 16 + rrow) * LDS_STRIDE + kc * 32 + koff);
#pragma unroll
      for (int g = 0; g < 4; ++g)
        acc[g] = wmma_bf16f32(a, bf[g], acc[g]);
    }
    cur ^= 1;
  }
#else
  // Fallback: cooperative global->VGPR->LDS fill, single buffer.
  const int frow  = tid >> 1;
  const int fhalf = tid & 1;
  const bf16_t* wrow = W + (size_t)((frow >> 4) * HID + n_base + (frow & 15)) * K
                         + fhalf * 64;
  bf16_t* lrow = smem + frow * LDS_STRIDE + fhalf * 64;

  for (int ci = 0; ci < nchunk; ++ci) {
    const int kk0 = ci * KCHUNK;
    __syncthreads();
#pragma unroll
    for (int s = 0; s < 8; ++s)
      *(v8bf*)(lrow + s * 8) = *(const v8bf*)(wrow + kk0 + s * 8);
    __syncthreads();
#pragma unroll
    for (int kc = 0; kc < 4; ++kc) {
      v16bf a = load_a16(arow, kk0 + kc * 32, khalf);
      v16bf bf[4];
#pragma unroll
      for (int g = 0; g < 4; ++g)
        bf[g] = load_b16_lds(smem + (g * 16 + rrow) * LDS_STRIDE + kc * 32 + koff);
#pragma unroll
      for (int g = 0; g < 4; ++g)
        acc[g] = wmma_bf16f32(a, bf[g], acc[g]);
    }
  }
#endif
}

// ---------------------------------------------------------------------------
// LSTM cell nonlinearity fused on WMMA accumulators.
// ---------------------------------------------------------------------------
__device__ __forceinline__ void cell_update(
    v8f acc[4], const float* __restrict__ bias, int n_base,
    float* __restrict__ cbuf, bf16_t* __restrict__ hbf,
    float* __restrict__ hf, int write_f32)
{
  const int lane   = threadIdx.x & 31;
  const int wave   = threadIdx.x >> 5;
  const int m_base = wave * 16;
  const int rrow   = lane & 15;
  const int madd   = (lane < 16) ? 0 : 8;
  const int n      = n_base + rrow;

  float bg[4];
#pragma unroll
  for (int g = 0; g < 4; ++g) bg[g] = bias[g * HID + n];

#pragma unroll
  for (int v = 0; v < 8; ++v) {
    const int m   = m_base + v + madd;
    const int idx = m * HID + n;
    float iv = sigmoidf_(acc[0][v] + bg[0]);
    float fv = sigmoidf_(acc[1][v] + bg[1]);
    float gv = tanhf(acc[2][v] + bg[2]);
    float ov = sigmoidf_(acc[3][v] + bg[3]);
    float cold = cbuf[idx];
    float cnew = fv * cold + iv * gv;
    cbuf[idx] = cnew;
    float hv = ov * tanhf(cnew);
    hbf[idx] = (bf16_t)hv;
    if (write_f32) hf[idx] = hv;
  }
}

// ---------------------------------------------------------------------------
// Persistent 2-layer LSTM: 64 blocks x 128 threads, entire SEQ loop on-chip,
// 2 grid barriers per timestep. Weights stay L2-resident (28MB << 192MB L2).
// ---------------------------------------------------------------------------
__global__ void __launch_bounds__(128)
lstm_persistent_kernel(
    const bf16_t* __restrict__ xs,                 // [SEQ, BATCH, EMB]
    const bf16_t* __restrict__ Wih0,               // [4H, EMB]
    const bf16_t* __restrict__ Whh0,               // [4H, HID]
    const float*  __restrict__ b0,                 // [4H]
    const bf16_t* __restrict__ Wih1,               // [4H, HID]
    const bf16_t* __restrict__ Whh1,               // [4H, HID]
    const float*  __restrict__ b1,                 // [4H]
    bf16_t* __restrict__ h0bf0, bf16_t* __restrict__ h0bf1,
    bf16_t* __restrict__ h1bf0, bf16_t* __restrict__ h1bf1,
    float* __restrict__ c0buf, float* __restrict__ c1buf,
    float* __restrict__ h0f,   float* __restrict__ h1f,
    unsigned* bar_cnt, unsigned* bar_gen)
{
  __shared__ __align__(256) bf16_t smem[2 * BUFE];
  const int n_base = blockIdx.x * 16;
  const v8f z = {0.f, 0.f, 0.f, 0.f, 0.f, 0.f, 0.f, 0.f};

  for (int t = 0; t < SEQ; ++t) {
    const bf16_t* h0cur = (t & 1) ? h0bf1 : h0bf0;
    bf16_t*       h0nxt = (t & 1) ? h0bf0 : h0bf1;
    const bf16_t* h1cur = (t & 1) ? h1bf1 : h1bf0;
    bf16_t*       h1nxt = (t & 1) ? h1bf0 : h1bf1;
    const int last = (t == SEQ - 1) ? 1 : 0;

    // ---- layer 0 ----
    v8f acc[4] = {z, z, z, z};
    gemm_part(acc, xs + (size_t)t * BATCH * EMB, EMB, Wih0, n_base, smem);
    gemm_part(acc, h0cur, HID, Whh0, n_base, smem);
    cell_update(acc, b0, n_base, c0buf, h0nxt, h0f, last);
    gsync(bar_cnt, bar_gen);               // h0[t] fully written

    // ---- layer 1 ----
    v8f acc2[4] = {z, z, z, z};
    gemm_part(acc2, h0nxt, HID, Wih1, n_base, smem);
    gemm_part(acc2, h1cur, HID, Whh1, n_base, smem);
    cell_update(acc2, b1, n_base, c1buf, h1nxt, h1f, last);
    gsync(bar_cnt, bar_gen);               // h1[t] fully written
  }
}

// ---------------------------------------------------------------------------
// Final projection: logits[64,128] = h_last @ Wfc^T + bfc  (WMMA, K=1024)
// ---------------------------------------------------------------------------
__global__ void fc_kernel(const bf16_t* __restrict__ h,      // [BATCH, HID]
                          const bf16_t* __restrict__ Wfc,    // [VOCAB, HID]
                          const float*  __restrict__ bfc,    // [VOCAB]
                          float*        __restrict__ out)    // [BATCH, VOCAB]
{
  const int lane   = threadIdx.x & 31;
  const int wave   = threadIdx.x >> 5;
  const int m_base = wave * 16;
  const int n_base = blockIdx.x * 16;
  const int rrow   = lane & 15;
  const int khalf  = (lane < 16) ? 0 : 8;
  const int koff   = (lane < 16) ? 0 : 16;

  v8f acc = {0.f, 0.f, 0.f, 0.f, 0.f, 0.f, 0.f, 0.f};
  const bf16_t* arow = h   + (size_t)(m_base + rrow) * HID;
  const bf16_t* brow = Wfc + (size_t)(n_base + rrow) * HID;

  for (int kk = 0; kk < HID; kk += 32) {
    union { v16bf v; v8bf hh[2]; } u;
    const bf16_t* p = brow + kk + koff;
    u.hh[0] = *(const v8bf*)(p);
    u.hh[1] = *(const v8bf*)(p + 8);
    acc = wmma_bf16f32(load_a16(arow, kk, khalf), u.v, acc);
  }

  const float bn = bfc[n_base + rrow];
  const int madd = (lane < 16) ? 0 : 8;
#pragma unroll
  for (int v = 0; v < 8; ++v) {
    const int m = m_base + v + madd;
    out[m * VOCAB + n_base + rrow] = acc[v] + bn;
  }
}

// ---------------------------------------------------------------------------
// Utility kernels
// ---------------------------------------------------------------------------
__global__ void f32_to_bf16_kernel(const float* __restrict__ src,
                                   bf16_t* __restrict__ dst, int n) {
  int i = blockIdx.x * blockDim.x + threadIdx.x;
  if (i < n) dst[i] = (bf16_t)src[i];
}

__global__ void embed_kernel(const int* __restrict__ x,
                             const float* __restrict__ emb,
                             bf16_t* __restrict__ xs, int total) {
  int i = blockIdx.x * blockDim.x + threadIdx.x;
  if (i >= total) return;
  int e  = i & (EMB - 1);
  int sb = i >> 9;                  // EMB == 512
  int tok = x[sb];
  xs[i] = (bf16_t)emb[tok * EMB + e];
}

__global__ void init_state_kernel(const float* __restrict__ h0,
                                  const float* __restrict__ c0,
                                  bf16_t* __restrict__ h0bf,
                                  bf16_t* __restrict__ h1bf,
                                  float* __restrict__ c0b,
                                  float* __restrict__ c1b,
                                  unsigned* bar_cnt, unsigned* bar_gen) {
  int i = blockIdx.x * blockDim.x + threadIdx.x;
  if (i == 0) { *bar_cnt = 0u; *bar_gen = 0u; }
  if (i >= BH) return;
  h0bf[i] = (bf16_t)h0[i];
  h1bf[i] = (bf16_t)h0[BH + i];
  c0b[i] = c0[i];
  c1b[i] = c0[BH + i];
}

__global__ void pack_kernel(const float* __restrict__ h0f,
                            const float* __restrict__ h1f,
                            const float* __restrict__ c0b,
                            const float* __restrict__ c1b,
                            float* __restrict__ out) {
  int i = blockIdx.x * blockDim.x + threadIdx.x;
  if (i >= BH) return;
  out[LOGITS + i]          = h0f[i];   // h stack, layer 0
  out[LOGITS + BH + i]     = h1f[i];   // h stack, layer 1
  out[LOGITS + 2 * BH + i] = c0b[i];   // c stack, layer 0
  out[LOGITS + 3 * BH + i] = c1b[i];   // c stack, layer 1
}

// ---------------------------------------------------------------------------
extern "C" void kernel_launch(void* const* d_in, const int* in_sizes, int n_in,
                              void* d_out, int out_size, void* d_ws, size_t ws_size,
                              hipStream_t stream) {
  (void)in_sizes; (void)n_in; (void)out_size; (void)ws_size;

  const int*   x    = (const int*)  d_in[0];
  const float* h0   = (const float*)d_in[1];
  const float* c0   = (const float*)d_in[2];
  const float* emb  = (const float*)d_in[3];
  const float* Wih0 = (const float*)d_in[4];
  const float* Whh0 = (const float*)d_in[5];
  const float* b0   = (const float*)d_in[6];
  const float* Wih1 = (const float*)d_in[7];
  const float* Whh1 = (const float*)d_in[8];
  const float* b1   = (const float*)d_in[9];
  const float* Wfc  = (const float*)d_in[10];
  const float* bfc  = (const float*)d_in[11];
  float* out = (float*)d_out;

  char* ws = (char*)d_ws;
  size_t off = 0;
  auto alloc = [&](size_t bytes) -> void* {
    void* p = ws + off;
    off += (bytes + 255) & ~(size_t)255;
    return p;
  };

  bf16_t* xs_bf   = (bf16_t*)alloc((size_t)SEQ * BATCH * EMB * sizeof(bf16_t));
  bf16_t* Wih0_bf = (bf16_t*)alloc((size_t)4 * HID * EMB * sizeof(bf16_t));
  bf16_t* Whh0_bf = (bf16_t*)alloc((size_t)4 * HID * HID * sizeof(bf16_t));
  bf16_t* Wih1_bf = (bf16_t*)alloc((size_t)4 * HID * HID * sizeof(bf16_t));
  bf16_t* Whh1_bf = (bf16_t*)alloc((size_t)4 * HID * HID * sizeof(bf16_t));
  bf16_t* Wfc_bf  = (bf16_t*)alloc((size_t)VOCAB * HID * sizeof(bf16_t));
  bf16_t* h0bf0   = (bf16_t*)alloc(BH * sizeof(bf16_t));
  bf16_t* h0bf1   = (bf16_t*)alloc(BH * sizeof(bf16_t));
  bf16_t* h1bf0   = (bf16_t*)alloc(BH * sizeof(bf16_t));
  bf16_t* h1bf1   = (bf16_t*)alloc(BH * sizeof(bf16_t));
  float* h0f = (float*)alloc(BH * sizeof(float));
  float* h1f = (float*)alloc(BH * sizeof(float));
  float* c0b = (float*)alloc(BH * sizeof(float));
  float* c1b = (float*)alloc(BH * sizeof(float));
  unsigned* bar_cnt = (unsigned*)alloc(sizeof(unsigned));
  unsigned* bar_gen = (unsigned*)alloc(sizeof(unsigned));

  const int T = 256;
  f32_to_bf16_kernel<<<(4 * HID * EMB + T - 1) / T, T, 0, stream>>>(Wih0, Wih0_bf, 4 * HID * EMB);
  f32_to_bf16_kernel<<<(4 * HID * HID + T - 1) / T, T, 0, stream>>>(Whh0, Whh0_bf, 4 * HID * HID);
  f32_to_bf16_kernel<<<(4 * HID * HID + T - 1) / T, T, 0, stream>>>(Wih1, Wih1_bf, 4 * HID * HID);
  f32_to_bf16_kernel<<<(4 * HID * HID + T - 1) / T, T, 0, stream>>>(Whh1, Whh1_bf, 4 * HID * HID);
  f32_to_bf16_kernel<<<(VOCAB * HID + T - 1) / T, T, 0, stream>>>(Wfc, Wfc_bf, VOCAB * HID);

  const int etotal = SEQ * BATCH * EMB;
  embed_kernel<<<(etotal + T - 1) / T, T, 0, stream>>>(x, emb, xs_bf, etotal);
  init_state_kernel<<<(BH + T - 1) / T, T, 0, stream>>>(h0, c0, h0bf0, h1bf0,
                                                        c0b, c1b, bar_cnt, bar_gen);

  lstm_persistent_kernel<<<NBLK, 128, 0, stream>>>(
      xs_bf, Wih0_bf, Whh0_bf, b0, Wih1_bf, Whh1_bf, b1,
      h0bf0, h0bf1, h1bf0, h1bf1, c0b, c1b, h0f, h1f, bar_cnt, bar_gen);

  // after t=SEQ-1 (odd), final h lives in the "0" buffers
  fc_kernel<<<VOCAB / 16, 128, 0, stream>>>(h1bf0, Wfc_bf, bfc, out);
  pack_kernel<<<(BH + T - 1) / T, T, 0, stream>>>(h0f, h1f, c0b, c1b, out);
}
